// TemporalRULPredictor_76725295775920
// MI455X (gfx1250) — compile-verified, matrix operand order checked by reading, not run on hardware
//
#include <hip/hip_runtime.h>
#include <hip/hip_bf16.h>
#include <stdint.h>

// ---------------------------------------------------------------------------
// MI455X (gfx1250) implementation: GATv2(x3) -> BiLSTM(x2) -> 3 RUL heads.
// All dense matmuls run through V_WMMA_F32_16X16X32_F16 (f16 in, f32 accum).
// ---------------------------------------------------------------------------

typedef _Float16 half_t;
typedef __attribute__((ext_vector_type(16))) _Float16 v16h;
typedef __attribute__((ext_vector_type(8)))  _Float16 v8h;
typedef __attribute__((ext_vector_type(8)))  float    v8f;

#define DEVINL __device__ __forceinline__
#define CDIV(a, b) (((a) + (b) - 1) / (b))

static constexpr int B_  = 1024;
static constexpr int T_  = 64;
static constexpr int NN  = 7;          // nodes per graph
static constexpr int FD  = 15;         // input features
static constexpr int NV  = B_ * NN;    // 7168 nodes total
static constexpr int TN  = T_ * NV;    // 458752 rows (all timesteps)
static constexpr int HD_ = 384;        // H*D
static constexpr int HID = 96;         // lstm hidden

// ---------------------------------------------------------------------------
// WMMA fragment load: 16-bit A/B matrix, 16x32 tile, row-major [row, K].
// lanes 0-15 : row = base+lane,  K = {k0..k0+7, k0+16..k0+23}
// lanes 16-31: row = base+lane-16, K = {k0+8..k0+15, k0+24..k0+31}
// ---------------------------------------------------------------------------
DEVINL v16h load_frag(const half_t* p) {
  v8h lo = *(const v8h*)(p);
  v8h hi = *(const v8h*)(p + 16);
  v16h r;
#pragma unroll
  for (int i = 0; i < 8; ++i) { r[i] = lo[i]; r[i + 8] = hi[i]; }
  return r;
}

DEVINL float sigm(float x) { return 1.f / (1.f + expf(-x)); }

DEVINL void atomicMaxF(float* addr, float val) {
  // monotone int encoding trick; correct for mixed-sign concurrent updates
  if (val >= 0.f) atomicMax((int*)addr, __float_as_int(val));
  else            atomicMin((unsigned int*)addr, __float_as_uint(val));
}

// ---------------------------------------------------------------------------
// WMMA GEMM: C[M,N] = A[M,K] * Bt[N,K]^T  (+ optional f32 C preload)
// A,Bt f16 row-major. One wave -> 16 x (NACT*16) tile, 8 waves/block.
// NACT is compile-time so all accumulators live in fixed even VGPR octets.
// Per k-step: phase 1 loads A + all NACT B fragments into distinct registers
// (one big load clause, latency overlapped), phase 2 bursts the NACT WMMAs.
// M mult of 16 (grid covers mult of 128), N = gridDim.x*NACT*16, K mult of 32.
// ---------------------------------------------------------------------------
template <int NACT, bool PRELOAD, bool OUT_HALF>
__global__ __launch_bounds__(256) void k_gemm(
    const half_t* __restrict__ A, const half_t* __restrict__ Bt,
    const float* __restrict__ Cpre, void* __restrict__ Cout,
    int M, int K, int lda, int ldb, int ldc) {
  const int wave = threadIdx.x >> 5;
  const int lane = threadIdx.x & 31;
  const int tn0  = blockIdx.x * (NACT * 16);
  const int tm   = blockIdx.y * 128 + wave * 16;
  if (tm >= M) return;                       // wave-uniform: EXEC stays full
  const int hoff = (lane & 16) ? 8 : 0;      // K half-offset (A/B), M half-offset (C)
  const int l15  = lane & 15;
  const half_t* aRow = A + (size_t)(tm + l15) * lda;

  const half_t* bRow[NACT];
#pragma unroll
  for (int j = 0; j < NACT; ++j)
    bRow[j] = Bt + (size_t)(tn0 + j * 16 + l15) * ldb + hoff;

  v8f acc[NACT];
#pragma unroll
  for (int j = 0; j < NACT; ++j) {
    if (PRELOAD) {
      const int col = tn0 + j * 16 + l15;
#pragma unroll
      for (int r = 0; r < 8; ++r)
        acc[j][r] = Cpre[(size_t)(tm + r + hoff) * ldc + col];
    } else {
#pragma unroll
      for (int r = 0; r < 8; ++r) acc[j][r] = 0.f;
    }
  }

  for (int k0 = 0; k0 < K; k0 += 32) {
    // phase 1: issue all fragment loads into distinct registers
    v16h a = load_frag(aRow + k0 + hoff);
    v16h bf[NACT];
#pragma unroll
    for (int j = 0; j < NACT; ++j) bf[j] = load_frag(bRow[j] + k0);
    __builtin_prefetch(aRow + k0 + 64, 0, 1);   // global_prefetch_b8
    // phase 2: burst the WMMAs
#pragma unroll
    for (int j = 0; j < NACT; ++j)
      acc[j] = __builtin_amdgcn_wmma_f32_16x16x32_f16(
          false, a, false, bf[j], (short)0, acc[j], false, false);
  }

#pragma unroll
  for (int j = 0; j < NACT; ++j) {
    const int col = tn0 + j * 16 + l15;
#pragma unroll
    for (int r = 0; r < 8; ++r) {
      size_t o = (size_t)(tm + r + hoff) * ldc + col;
      if (OUT_HALF) ((half_t*)Cout)[o] = (half_t)acc[j][r];
      else          ((float*)Cout)[o]  = acc[j][r];
    }
  }
}

// ---------------------------------------------------------------------------
// Weight conversion kernels (run once per launch; tiny)
// ---------------------------------------------------------------------------
// transpose + pad: src f32 [K,N] -> dst f16 [Npad, Kpad]
__global__ void k_wT(const float* __restrict__ src, half_t* __restrict__ dst,
                     int K, int N, int Kpad, int Npad) {
  int idx = blockIdx.x * 256 + threadIdx.x;
  if (idx >= Npad * Kpad) return;
  int kk = idx % Kpad, nn = idx / Kpad;
  float v = (kk < K && nn < N) ? src[(size_t)kk * N + nn] : 0.f;
  dst[idx] = (half_t)v;
}
// direct convert: src f32 [N,K] -> dst f16 [N,K]
__global__ void k_wD(const float* __restrict__ src, half_t* __restrict__ dst, int count) {
  int idx = blockIdx.x * 256 + threadIdx.x;
  if (idx < count) dst[idx] = (half_t)src[idx];
}

// x_sequence [B,T,N,FD] f32 -> x0 [T*NV, 32] f16 (K padded 15->32)
__global__ void k_conv_x0(const float* __restrict__ x, half_t* __restrict__ x0) {
  int idx = blockIdx.x * 256 + threadIdx.x;
  if (idx >= TN * 32) return;
  int col = idx & 31;
  int row = idx >> 5;
  int t = row / NV;
  int r = row - t * NV;
  int b = r / NN;
  int nd = r - b * NN;
  float v = 0.f;
  if (col < FD) v = x[(((size_t)b * T_ + t) * NN + nd) * FD + col];
  x0[idx] = (half_t)v;
}

__global__ void k_fill_f(float* p, float v, int count) {
  int idx = blockIdx.x * 256 + threadIdx.x;
  if (idx < count) p[idx] = v;
}
__global__ void k_fill_h(half_t* p, int count) {
  int idx = blockIdx.x * 256 + threadIdx.x;
  if (idx < count) p[idx] = (half_t)0.f;
}
__global__ void k_bias_init(float* __restrict__ out, const float* __restrict__ bias, int rows) {
  int idx = blockIdx.x * 256 + threadIdx.x;
  if (idx >= rows * HD_) return;
  out[idx] = bias[idx % HD_];
}

// ---------------------------------------------------------------------------
// GAT edge kernels.  xlr f16 [T*NV, 768] = [xl | xr].
// wave per (t, edge, head): e = sum_d leaky(xl[src]+xr[dst]) * att; segment max
// ---------------------------------------------------------------------------
__global__ __launch_bounds__(256) void k_logits(
    const half_t* __restrict__ xlr, const long long* __restrict__ ei,
    const float* __restrict__ att, float* __restrict__ ebuf,
    float* __restrict__ mbuf, int E, int Ep) {
  int wid  = blockIdx.x * 8 + (threadIdx.x >> 5);
  int lane = threadIdx.x & 31;
  int total = T_ * Ep * 4;
  if (wid >= total) return;
  int t   = wid / (Ep * 4);
  int rem = wid - t * (Ep * 4);
  int e = rem >> 2, h = rem & 3;
  int sn = (e < E) ? (int)ei[e]     : (e - E);
  int dn = (e < E) ? (int)ei[E + e] : (e - E);
  size_t srow = (size_t)(t * NV + sn) * 768;
  size_t drow = (size_t)(t * NV + dn) * 768 + 384;
  float acc = 0.f;
#pragma unroll
  for (int k = 0; k < 3; ++k) {
    int dd = h * 96 + k * 32 + lane;
    float v = (float)xlr[srow + dd] + (float)xlr[drow + dd];
    v = v > 0.f ? v : 0.2f * v;              // leaky_relu(0.2)
    acc += v * att[dd];
  }
#pragma unroll
  for (int o = 16; o; o >>= 1) acc += __shfl_xor(acc, o, 32);
  if (lane == 0) {
    ebuf[(size_t)(t * Ep + e) * 4 + h] = acc;
    atomicMaxF(&mbuf[(size_t)(t * NV + dn) * 4 + h], acc);
  }
}

// w = exp(e - m[dst]); z[dst] += w   (w overwrites e)
__global__ void k_expsum(float* __restrict__ ebuf, const float* __restrict__ mbuf,
                         float* __restrict__ zbuf, const long long* __restrict__ ei,
                         int E, int Ep) {
  int idx = blockIdx.x * 256 + threadIdx.x;
  int total = T_ * Ep * 4;
  if (idx >= total) return;
  int h = idx & 3;
  int rem = idx >> 2;      // t*Ep + e
  int e = rem % Ep;
  int t = rem / Ep;
  int dn = (e < E) ? (int)ei[E + e] : (e - E);
  float w = expf(ebuf[idx] - mbuf[(size_t)(t * NV + dn) * 4 + h]);
  ebuf[idx] = w;
  atomicAdd(&zbuf[(size_t)(t * NV + dn) * 4 + h], w);
}

// agg[dst, d] += xl[src, d] * (w / z[dst])
__global__ void k_agg(const half_t* __restrict__ xlr, const float* __restrict__ ebuf,
                      const float* __restrict__ zbuf, const long long* __restrict__ ei,
                      float* __restrict__ agg, int E, int Ep) {
  int idx = blockIdx.x * 256 + threadIdx.x;
  int total = T_ * Ep * HD_;          // ~1.23e9, fits int32
  if (idx >= total) return;
  int d   = idx % HD_;
  int rem = idx / HD_;
  int e = rem % Ep;
  int t = rem / Ep;
  int h = d / 96;
  int sn = (e < E) ? (int)ei[e]     : (e - E);
  int dn = (e < E) ? (int)ei[E + e] : (e - E);
  float alpha = ebuf[(size_t)(t * Ep + e) * 4 + h] /
                zbuf[(size_t)(t * NV + dn) * 4 + h];
  float v = (float)xlr[(size_t)(t * NV + sn) * 768 + d] * alpha;
  atomicAdd(&agg[(size_t)(t * NV + dn) * HD_ + d], v);
}

// LayerNorm(384) + ELU, f32 in -> f16 out. One wave per row (12 elems/lane).
__global__ __launch_bounds__(256) void k_lnelu(
    const float* __restrict__ X, const float* __restrict__ g,
    const float* __restrict__ b, half_t* __restrict__ out, int rows) {
  int wid  = blockIdx.x * 8 + (threadIdx.x >> 5);
  int lane = threadIdx.x & 31;
  if (wid >= rows) return;
  const float* x = X + (size_t)wid * HD_;
  float s = 0.f, ss = 0.f;
#pragma unroll
  for (int k = 0; k < 12; ++k) { float v = x[k * 32 + lane]; s += v; ss += v * v; }
#pragma unroll
  for (int o = 16; o; o >>= 1) { s += __shfl_xor(s, o, 32); ss += __shfl_xor(ss, o, 32); }
  float mu   = s * (1.f / HD_);
  float var  = ss * (1.f / HD_) - mu * mu;
  float rstd = rsqrtf(var + 1e-5f);
  half_t* y = out + (size_t)wid * HD_;
#pragma unroll
  for (int k = 0; k < 12; ++k) {
    int d = k * 32 + lane;
    float v = (x[d] - mu) * rstd * g[d] + b[d];
    v = v > 0.f ? v : expm1f(v);             // ELU
    y[d] = (half_t)v;
  }
}

// ---------------------------------------------------------------------------
// LSTM pointwise step.  gates = gatesx[t] + h@Whh^T (already summed by WMMA
// C-preload).  torch gate order i,f,g,o.
// ---------------------------------------------------------------------------
__global__ void k_lstm_step(const float* __restrict__ gates, const float* __restrict__ b,
                            float* __restrict__ c, half_t* __restrict__ h,
                            half_t* __restrict__ outc, int t, int dirOff) {
  int idx = blockIdx.x * 256 + threadIdx.x;
  if (idx >= NV * HID) return;
  int node = idx / HID, j = idx - node * HID;
  const float* g = gates + (size_t)node * 384;
  float gi = g[j]       + b[j];
  float gf = g[96 + j]  + b[96 + j];
  float gg = g[192 + j] + b[192 + j];
  float go = g[288 + j] + b[288 + j];
  float cc = sigm(gf) * c[idx] + sigm(gi) * tanhf(gg);
  float hh = sigm(go) * tanhf(cc);
  c[idx] = cc;
  h[idx] = (half_t)hh;
  outc[((size_t)t * NV + node) * 192 + dirOff + j] = (half_t)hh;
}

// relu(x + b1) then LayerNorm(96) -> f16.  One wave per row (3 elems/lane).
__global__ __launch_bounds__(256) void k_relu_ln(
    const float* __restrict__ X, const float* __restrict__ b1,
    const float* __restrict__ g, const float* __restrict__ bb,
    half_t* __restrict__ out, int rows) {
  int wid  = blockIdx.x * 8 + (threadIdx.x >> 5);
  int lane = threadIdx.x & 31;
  if (wid >= rows) return;
  const float* x = X + (size_t)wid * 96;
  float v[3]; float s = 0.f, ss = 0.f;
#pragma unroll
  for (int k = 0; k < 3; ++k) {
    int d = k * 32 + lane;
    float t = x[d] + b1[d];
    t = t > 0.f ? t : 0.f;
    v[k] = t; s += t; ss += t * t;
  }
#pragma unroll
  for (int o = 16; o; o >>= 1) { s += __shfl_xor(s, o, 32); ss += __shfl_xor(ss, o, 32); }
  float mu   = s * (1.f / 96.f);
  float var  = ss * (1.f / 96.f) - mu * mu;
  float rstd = rsqrtf(var + 1e-5f);
#pragma unroll
  for (int k = 0; k < 3; ++k) {
    int d = k * 32 + lane;
    out[(size_t)wid * 96 + d] = (half_t)((v[k] - mu) * rstd * g[d] + bb[d]);
  }
}

// relu(x + bias[col%96]) -> f16
__global__ void k_relu96(const float* __restrict__ X, const float* __restrict__ bias,
                         half_t* __restrict__ out, int count) {
  int idx = blockIdx.x * 256 + threadIdx.x;
  if (idx >= count) return;
  float t = X[idx] + bias[idx % 96];
  out[idx] = (half_t)(t > 0.f ? t : 0.f);
}

// diagonal extract + sigmoid / softplus -> d_out [2,3,B,N]
__global__ void k_extract(const float* __restrict__ r2, const float* __restrict__ u2,
                          const float* __restrict__ b2, const float* __restrict__ c2,
                          float* __restrict__ out, int head) {
  int idx = blockIdx.x * 256 + threadIdx.x;
  if (idx >= NV) return;
  int node = idx % NN;
  float rv = r2[(size_t)idx * 16 + node] + b2[node];
  float uv = u2[(size_t)idx * 16 + node] + c2[node];
  rv = 1.f / (1.f + expf(-rv));
  uv = (uv > 20.f) ? uv : log1pf(expf(uv));
  out[(size_t)(0 * 3 + head) * NV + idx] = rv;
  out[(size_t)(1 * 3 + head) * NV + idx] = uv;
}

// ---------------------------------------------------------------------------
extern "C" void kernel_launch(void* const* d_in, const int* in_sizes, int n_in,
                              void* d_out, int out_size, void* d_ws, size_t ws_size,
                              hipStream_t stream) {
  const float*     xseq = (const float*)d_in[0];
  const long long* ei   = (const long long*)d_in[1];
  const int E  = in_sizes[1] / 2;
  const int Ep = E + NV;

  const float *gWl[3], *gWr[3], *gAtt[3], *gBias[3], *gLg[3], *gLb[3];
  for (int i = 0; i < 3; ++i) {
    const int b = 2 + i * 6;
    gWl[i]   = (const float*)d_in[b];     gWr[i] = (const float*)d_in[b + 1];
    gAtt[i]  = (const float*)d_in[b + 2]; gBias[i] = (const float*)d_in[b + 3];
    gLg[i]   = (const float*)d_in[b + 4]; gLb[i] = (const float*)d_in[b + 5];
  }
  const float *Wih[2][2], *Whh[2][2], *bL[2][2];
  for (int l = 0; l < 2; ++l) {
    const int b = 20 + l * 6;
    Wih[l][0] = (const float*)d_in[b];     Whh[l][0] = (const float*)d_in[b + 1];
    bL[l][0]  = (const float*)d_in[b + 2];
    Wih[l][1] = (const float*)d_in[b + 3]; Whh[l][1] = (const float*)d_in[b + 4];
    bL[l][1]  = (const float*)d_in[b + 5];
  }
  const float *hW1[3], *hb1[3], *hLg[3], *hLb[3], *hW2[3], *hb2[3],
              *hU1[3], *hc1[3], *hU2[3], *hc2[3];
  for (int i = 0; i < 3; ++i) {
    const int b = 32 + i * 10;
    hW1[i] = (const float*)d_in[b];     hb1[i] = (const float*)d_in[b + 1];
    hLg[i] = (const float*)d_in[b + 2]; hLb[i] = (const float*)d_in[b + 3];
    hW2[i] = (const float*)d_in[b + 4]; hb2[i] = (const float*)d_in[b + 5];
    hU1[i] = (const float*)d_in[b + 6]; hc1[i] = (const float*)d_in[b + 7];
    hU2[i] = (const float*)d_in[b + 8]; hc2[i] = (const float*)d_in[b + 9];
  }

  // ---- workspace carve -----------------------------------------------------
  char* wsb = (char*)d_ws; size_t off = 0;
  auto carve = [&](size_t bytes) -> void* {
    void* p = wsb + off; off = (off + bytes + 255) & ~(size_t)255; return p;
  };
  half_t* x0    = (half_t*)carve((size_t)TN * 32 * 2);
  half_t* emb   = (half_t*)carve((size_t)TN * HD_ * 2);
  half_t* xlr   = (half_t*)carve((size_t)TN * 768 * 2);
  float*  agg   = (float*) carve((size_t)TN * HD_ * 4);
  float*  ebuf  = (float*) carve((size_t)T_ * Ep * 4 * 4);
  float*  mbuf  = (float*) carve((size_t)T_ * NV * 4 * 4);
  float*  zbuf  = (float*) carve((size_t)T_ * NV * 4 * 4);
  half_t* lout1 = (half_t*)carve((size_t)TN * 192 * 2);
  half_t* lout2 = (half_t*)carve((size_t)TN * 192 * 2);
  float*  gtmp  = (float*) carve((size_t)NV * 384 * 4);
  half_t* hsv   = (half_t*)carve((size_t)NV * HID * 2);
  float*  cst   = (float*) carve((size_t)NV * HID * 4);
  float*  h1f   = (float*) carve((size_t)NV * 96 * 4);
  half_t* h1h   = (half_t*)carve((size_t)NV * 96 * 2);
  float*  r2    = (float*) carve((size_t)NV * 16 * 4);
  float*  u2    = (float*) carve((size_t)NV * 16 * 4);
  // weights (f16)
  half_t* wlr[3];
  wlr[0] = (half_t*)carve((size_t)768 * 32 * 2);
  wlr[1] = (half_t*)carve((size_t)768 * 384 * 2);
  wlr[2] = (half_t*)carve((size_t)768 * 384 * 2);
  half_t *wihC[2][2], *whhC[2][2];
  for (int l = 0; l < 2; ++l)
    for (int d = 0; d < 2; ++d) {
      wihC[l][d] = (half_t*)carve((size_t)384 * (l ? 192 : 384) * 2);
      whhC[l][d] = (half_t*)carve((size_t)384 * 96 * 2);
    }
  half_t *w1t[3], *u1t[3], *w2t[3], *u2t[3];
  for (int i = 0; i < 3; ++i) {
    w1t[i] = (half_t*)carve((size_t)96 * 192 * 2);
    u1t[i] = (half_t*)carve((size_t)96 * 192 * 2);
    w2t[i] = (half_t*)carve((size_t)16 * 96 * 2);
    u2t[i] = (half_t*)carve((size_t)16 * 96 * 2);
  }
  // aliases: GAT scratch reused as LSTM gates_x (same byte size)
  float* gxf = agg;           // TN*384 f32
  float* gxr = (float*)xlr;   // TN*768 f16 == TN*384 f32 bytes, used after GAT only

  // ---- weight conversion ---------------------------------------------------
  for (int L = 0; L < 3; ++L) {
    int K = L ? 384 : FD, Kp = L ? 384 : 32;
    k_wT<<<CDIV(384 * Kp, 256), 256, 0, stream>>>(gWl[L], wlr[L],            K, HD_, Kp, 384);
    k_wT<<<CDIV(384 * Kp, 256), 256, 0, stream>>>(gWr[L], wlr[L] + 384 * Kp, K, HD_, Kp, 384);
  }
  for (int l = 0; l < 2; ++l)
    for (int d = 0; d < 2; ++d) {
      int Kin = l ? 192 : 384;
      k_wD<<<CDIV(384 * Kin, 256), 256, 0, stream>>>(Wih[l][d], wihC[l][d], 384 * Kin);
      k_wD<<<CDIV(384 * 96, 256), 256, 0, stream>>>(Whh[l][d], whhC[l][d], 384 * 96);
    }
  for (int i = 0; i < 3; ++i) {
    k_wT<<<CDIV(96 * 192, 256), 256, 0, stream>>>(hW1[i], w1t[i], 192, 96, 192, 96);
    k_wT<<<CDIV(96 * 192, 256), 256, 0, stream>>>(hU1[i], u1t[i], 192, 96, 192, 96);
    k_wT<<<CDIV(16 * 96, 256), 256, 0, stream>>>(hW2[i], w2t[i], 96, NN, 96, 16);
    k_wT<<<CDIV(16 * 96, 256), 256, 0, stream>>>(hU2[i], u2t[i], 96, NN, 96, 16);
  }

  // ---- GAT: 3 layers, all 64 timesteps batched -----------------------------
  k_conv_x0<<<CDIV(TN * 32, 256), 256, 0, stream>>>(xseq, x0);
  for (int L = 0; L < 3; ++L) {
    const half_t* Ain = L ? emb : x0;
    const int K = L ? 384 : 32;
    // [xl|xr] = X @ [Wl|Wr]   (M=458752, N=768, K)
    k_gemm<4, false, true><<<dim3(12, TN / 128), 256, 0, stream>>>(
        Ain, wlr[L], nullptr, xlr, TN, K, K, K, 768);
    k_bias_init<<<CDIV(TN * HD_, 256), 256, 0, stream>>>(agg, gBias[L], TN);
    k_fill_f<<<CDIV(T_ * NV * 4, 256), 256, 0, stream>>>(mbuf, -1e30f, T_ * NV * 4);
    k_fill_f<<<CDIV(T_ * NV * 4, 256), 256, 0, stream>>>(zbuf, 0.f, T_ * NV * 4);
    const int totW = T_ * Ep * 4;
    k_logits<<<CDIV(totW, 8), 256, 0, stream>>>(xlr, ei, gAtt[L], ebuf, mbuf, E, Ep);
    k_expsum<<<CDIV(totW, 256), 256, 0, stream>>>(ebuf, mbuf, zbuf, ei, E, Ep);
    k_agg<<<CDIV(T_ * Ep * HD_, 256), 256, 0, stream>>>(xlr, ebuf, zbuf, ei, agg, E, Ep);
    k_lnelu<<<CDIV(TN, 8), 256, 0, stream>>>(agg, gLg[L], gLb[L], emb, TN);
  }

  // ---- BiLSTM: 2 layers ----------------------------------------------------
  for (int l = 0; l < 2; ++l) {
    const half_t* Ain = l ? lout1 : emb;
    const int Kin = l ? 192 : 384;
    half_t* outc = l ? lout2 : lout1;
    // gates_x = X @ Wih^T for all timesteps (big batched WMMA GEMM, N=384)
    k_gemm<4, false, false><<<dim3(6, TN / 128), 256, 0, stream>>>(
        Ain, wihC[l][0], nullptr, gxf, TN, Kin, Kin, Kin, 384);
    k_gemm<4, false, false><<<dim3(6, TN / 128), 256, 0, stream>>>(
        Ain, wihC[l][1], nullptr, gxr, TN, Kin, Kin, Kin, 384);
    for (int d = 0; d < 2; ++d) {
      float* gx = d ? gxr : gxf;
      k_fill_h<<<CDIV(NV * HID, 256), 256, 0, stream>>>(hsv, NV * HID);
      k_fill_f<<<CDIV(NV * HID, 256), 256, 0, stream>>>(cst, 0.f, NV * HID);
      for (int s = 0; s < T_; ++s) {
        const int t = d ? (T_ - 1 - s) : s;
        // gates = gates_x[t] (C-preload) + h @ Whh^T   (M=7168,N=384,K=96)
        k_gemm<4, true, false><<<dim3(6, NV / 128), 256, 0, stream>>>(
            hsv, whhC[l][d], gx + (size_t)t * NV * 384, gtmp, NV, 96, 96, 96, 384);
        k_lstm_step<<<CDIV(NV * HID, 256), 256, 0, stream>>>(
            gtmp, bL[l][d], cst, hsv, outc, t, d * 96);
      }
    }
  }

  // ---- RUL heads on final timestep ----------------------------------------
  const half_t* finalA = lout2 + (size_t)(T_ - 1) * NV * 192;
  for (int hd = 0; hd < 3; ++hd) {
    // rul path: sigmoid(LN(relu(x@W1+b1)) @ W2 + b2)   (N=96 -> NACT=2, grid.x=3)
    k_gemm<2, false, false><<<dim3(3, NV / 128), 256, 0, stream>>>(
        finalA, w1t[hd], nullptr, h1f, NV, 192, 192, 192, 96);
    k_relu_ln<<<CDIV(NV, 8), 256, 0, stream>>>(h1f, hb1[hd], hLg[hd], hLb[hd], h1h, NV);
    k_gemm<1, false, false><<<dim3(1, NV / 128), 256, 0, stream>>>(
        h1h, w2t[hd], nullptr, r2, NV, 96, 96, 96, 16);
    // unc path: softplus(relu(x@U1+c1) @ U2 + c2)
    k_gemm<2, false, false><<<dim3(3, NV / 128), 256, 0, stream>>>(
        finalA, u1t[hd], nullptr, h1f, NV, 192, 192, 192, 96);
    k_relu96<<<CDIV(NV * 96, 256), 256, 0, stream>>>(h1f, hc1[hd], h1h, NV * 96);
    k_gemm<1, false, false><<<dim3(1, NV / 128), 256, 0, stream>>>(
        h1h, u2t[hd], nullptr, u2, NV, 96, 96, 96, 16);
    k_extract<<<CDIV(NV, 256), 256, 0, stream>>>(r2, u2, hb2[hd], hc2[hd],
                                                 (float*)d_out, hd);
  }
  (void)n_in; (void)out_size; (void)ws_size; (void)in_sizes;
}
